// Transformer_82918638616735
// MI455X (gfx1250) — matrix operand order repaired and checked by reading, not compiled, not run on hardware
//
#include <hip/hip_runtime.h>
#include <hip/hip_bf16.h>

// ---------------------------------------------------------------------------
// Transformer forward (encoder-decoder), BF16 WMMA on gfx1250 (MI455X).
// E=512, H=8, D=64, L=6, B=8, S=512, VOCAB=32000.
// GEMMs: v_wmma_f32_16x16x32_bf16; weight tiles staged to LDS via
// global_load_async_to_lds_b128 (ASYNCcnt) from a pre-converted bf16 copy.
// ---------------------------------------------------------------------------

#define NB   8
#define NS   512
#define NE   512
#define NH   8
#define ND   64
#define NL   6
#define NFF  2048
#define NV   32000
#define MTOK (NB * NS)          // 4096 token rows

typedef __attribute__((ext_vector_type(16))) __bf16        bf16x16;
typedef __attribute__((ext_vector_type(8)))  float         f32x8;
typedef __attribute__((ext_vector_type(4)))  unsigned int  u32x4;

union Frag { bf16x16 v; u32x4 q[2]; };

__device__ inline unsigned short f2bf(float f) {
    unsigned u = __float_as_uint(f);
    return (unsigned short)((u + 0x7FFFu + ((u >> 16) & 1u)) >> 16);   // RNE
}
__device__ inline float bf2f(unsigned short h) {
    return __uint_as_float(((unsigned)h) << 16);
}
__device__ inline unsigned pack2(float lo, float hi) {
    return (unsigned)f2bf(lo) | ((unsigned)f2bf(hi) << 16);
}

__device__ inline void wait_async0() {
#if __has_builtin(__builtin_amdgcn_s_wait_asynccnt)
    __builtin_amdgcn_s_wait_asynccnt(0);
#else
    asm volatile("s_wait_asynccnt 0x0" ::: "memory");
#endif
}

__device__ inline float warp_sum(float v) {
    #pragma unroll
    for (int o = 16; o > 0; o >>= 1) v += __shfl_xor(v, o, 32);
    return v;
}
__device__ inline float warp_max(float v) {
    #pragma unroll
    for (int o = 16; o > 0; o >>= 1) v = fmaxf(v, __shfl_xor(v, o, 32));
    return v;
}
__device__ inline float block_sum(float v, float* sh) {   // 256 threads = 8 waves
    v = warp_sum(v);
    int w = threadIdx.x >> 5;
    if ((threadIdx.x & 31) == 0) sh[w] = v;
    __syncthreads();
    float t = 0.f;
    #pragma unroll
    for (int i = 0; i < 8; ++i) t += sh[i];
    __syncthreads();
    return t;
}
__device__ inline float block_max(float v, float* sh) {
    v = warp_max(v);
    int w = threadIdx.x >> 5;
    if ((threadIdx.x & 31) == 0) sh[w] = v;
    __syncthreads();
    float t = -3.0e38f;
    #pragma unroll
    for (int i = 0; i < 8; ++i) t = fmaxf(t, sh[i]);
    __syncthreads();
    return t;
}

// f32 -> bf16 bulk convert (for weight matrices), 4 elems/thread
__global__ __launch_bounds__(256) void cvt_bf16_kernel(
    const float* __restrict__ src, unsigned short* __restrict__ dst, int n4)
{
    int i = blockIdx.x * 256 + threadIdx.x;
    if (i < n4) {
        float4 f = ((const float4*)src)[i];
        unsigned* d = (unsigned*)(void*)(dst + (size_t)i * 4);
        d[0] = pack2(f.x, f.y);
        d[1] = pack2(f.z, f.w);
    }
}

// ---------------------------------------------------------------------------
// GEMM: C[M,N] = A[M,K] * WB[N,K]^T + bias[N]  (optional ReLU)
// A: f32 (converted to bf16 while staging); WB: bf16, staged to LDS with
// global_load_async_to_lds_b128. Block 256 thr, tile 128x128x32, double buf,
// wave-grid 4(M) x 2(N), each wave 32x64 = 2x4 WMMA tiles.
// BOUND=false: M known multiple of 128 (no predication in hot loop).
// ---------------------------------------------------------------------------
#define BM 128
#define BN 128
#define BK 32
#define LDT 40   // padded LDS row stride in bf16 elems (80 B, 16B aligned)

template <bool BOUND>
__global__ __launch_bounds__(256) void gemm_bias_kernel(
    const float* __restrict__ A, const unsigned short* __restrict__ WB,
    const float* __restrict__ bias, float* __restrict__ C,
    int M, int N, int K, int relu)
{
    __shared__ unsigned short As[2][BM * LDT];
    __shared__ unsigned short Bs[2][BN * LDT];

    const int tid  = threadIdx.x;
    const int lane = tid & 31;
    const int wave = tid >> 5;
    const int wm   = wave & 3;          // 4 waves along M
    const int wn   = wave >> 2;         // 2 waves along N
    const int hlf  = lane >> 4;
    const int l16  = lane & 15;
    const int bm0  = blockIdx.y * BM;
    const int bn0  = blockIdx.x * BN;

    f32x8 acc[2][4] = {};
    const int nk = K / BK;

    // stage k-slice k0 into LDS buffer `buf`
    auto stage = [&](int k0, int buf) {
        // ---- B tile: async DMA global(bf16) -> LDS, 2 x b128 per thread ----
        for (int i = tid; i < BN * 4; i += 256) {
            int r = i >> 2, c8 = (i & 3) << 3;               // 8 bf16 per chunk
            const unsigned short* gp = WB + (size_t)(bn0 + r) * K + k0 + c8;
            unsigned lds = (unsigned)(uintptr_t)(void*)&Bs[buf][r * LDT + c8];
            asm volatile("global_load_async_to_lds_b128 %0, %1, off"
                         :: "v"(lds), "v"(gp) : "memory");
        }
        // ---- A tile: f32 -> bf16 through VGPRs ----
        for (int i = tid; i < BM * 8; i += 256) {            // 128 rows x 8 float4
            int r = i >> 3, c4 = (i & 7) << 2;
            int gr = bm0 + r;
            float4 f;
            if (BOUND) {
                f = make_float4(0.f, 0.f, 0.f, 0.f);
                if (gr < M) f = *(const float4*)(A + (size_t)gr * K + k0 + c4);
            } else {
                f = *(const float4*)(A + (size_t)gr * K + k0 + c4);
            }
            unsigned* dst = (unsigned*)(void*)&As[buf][r * LDT + c4];
            dst[0] = pack2(f.x, f.y);
            dst[1] = pack2(f.z, f.w);
        }
    };

    stage(0, 0);
    wait_async0();
    __syncthreads();
    int buf = 0;
    for (int kt = 0; kt < nk; ++kt) {
        if (kt + 1 < nk) stage((kt + 1) * BK, buf ^ 1);

        Frag af[2], bfrag[4];
        #pragma unroll
        for (int i = 0; i < 2; ++i) {
            const unsigned short* base = &As[buf][(wm * 32 + i * 16 + l16) * LDT];
            af[i].q[0] = *(const u32x4*)(const void*)(base + hlf * 8);
            af[i].q[1] = *(const u32x4*)(const void*)(base + 16 + hlf * 8);
        }
        #pragma unroll
        for (int j = 0; j < 4; ++j) {
            const unsigned short* base = &Bs[buf][(wn * 64 + j * 16 + l16) * LDT];
            bfrag[j].q[0] = *(const u32x4*)(const void*)(base + hlf * 8);
            bfrag[j].q[1] = *(const u32x4*)(const void*)(base + 16 + hlf * 8);
        }
        #pragma unroll
        for (int i = 0; i < 2; ++i)
            #pragma unroll
            for (int j = 0; j < 4; ++j)
                acc[i][j] = __builtin_amdgcn_wmma_f32_16x16x32_bf16(
                    false, af[i].v, false, bfrag[j].v,
                    (short)0, acc[i][j], false, false);

        wait_async0();       // our wave's async B-tile writes for buf^1 done
        __syncthreads();     // publish to all waves
        buf ^= 1;
    }

    // epilogue: C layout row m = v + 8*half, col n = lane&15
    #pragma unroll
    for (int i = 0; i < 2; ++i) {
        #pragma unroll
        for (int j = 0; j < 4; ++j) {
            int gn = bn0 + wn * 64 + j * 16 + l16;
            float bv = bias[gn];
            #pragma unroll
            for (int v = 0; v < 8; ++v) {
                int gm = bm0 + wm * 32 + i * 16 + hlf * 8 + v;
                float val = acc[i][j][v] + bv;
                if (relu) val = fmaxf(val, 0.f);
                if (!BOUND || gm < M) C[(size_t)gm * N + gn] = val;
            }
        }
    }
}

// ---------------------------------------------------------------------------
// Fused attention for one (b, h, 32-query block). Block 64 thr (2 waves),
// each wave owns 16 query rows. QH/KH: bf16 [B,H,S,D]; VT: bf16 [B,H,D,S].
// Mask faithful to the reference quirk: allowed iff k >= q when causal.
// Output written in concat layout OUT[b, q, h*D + d] (f32).
// ---------------------------------------------------------------------------
#define QB 32
#define PLD 520   // scores LDS row stride (bf16 elems)

__global__ __launch_bounds__(64) void attn_kernel(
    const unsigned short* __restrict__ QH,
    const unsigned short* __restrict__ KH,
    const unsigned short* __restrict__ VT,
    float* __restrict__ OUT, int causal)
{
    __shared__ unsigned short P[QB][PLD];    // scores -> probabilities (bf16)
    __shared__ unsigned short Qs[QB][72];    // staged Q block (bf16)
    __shared__ float red2[QB][2];
    __shared__ float rowsum[QB];

    const int tid  = threadIdx.x;
    const int lane = tid & 31;
    const int wave = tid >> 5;
    const int hlf  = lane >> 4;
    const int l16  = lane & 15;
    const int b    = blockIdx.z;
    const int h    = blockIdx.y;
    const int q0   = blockIdx.x * QB;
    const int hb   = b * NH + h;

    // stage Q tile (32 x 64 bf16)
    const unsigned short* Qp = QH + ((size_t)(hb * NS + q0)) * ND;
    for (int i = tid; i < QB * 8; i += 64) {
        int r = i >> 3, c = (i & 7) << 3;
        *(u32x4*)(void*)&Qs[r][c] = *(const u32x4*)(const void*)(Qp + r * ND + c);
    }
    __syncthreads();

    const int qr0 = wave * 16;

    // -------- phase 1: S = Q K^T * (1/sqrt(D)) with mask ----------
    const unsigned short* Kp = KH + (size_t)hb * NS * ND;
    for (int kt = 0; kt < NS / 16; ++kt) {
        f32x8 s = {};
        #pragma unroll
        for (int ks = 0; ks < 2; ++ks) {                 // D=64 -> two K=32 steps
            Frag a, bb;
            const unsigned short* ab = &Qs[qr0 + l16][ks * 32];
            a.q[0] = *(const u32x4*)(const void*)(ab + hlf * 8);
            a.q[1] = *(const u32x4*)(const void*)(ab + 16 + hlf * 8);
            const unsigned short* bp = Kp + (size_t)(kt * 16 + l16) * ND + ks * 32;
            bb.q[0] = *(const u32x4*)(const void*)(bp + hlf * 8);
            bb.q[1] = *(const u32x4*)(const void*)(bp + 16 + hlf * 8);
            s = __builtin_amdgcn_wmma_f32_16x16x32_bf16(
                    false, a.v, false, bb.v, (short)0, s, false, false);
        }
        #pragma unroll
        for (int v = 0; v < 8; ++v) {
            int m  = hlf * 8 + v;
            int qa = q0 + qr0 + m;
            int ka = kt * 16 + l16;
            float val = s[v] * 0.125f;                   // 1/sqrt(64)
            if (causal && (ka < qa)) val = -1.25e19f;    // (-1e20)*scale
            P[qr0 + m][ka] = f2bf(val);
        }
    }
    __syncthreads();

    // -------- phase 2: row softmax over 512 keys ----------
    {
        int r = tid >> 1, hh = tid & 1;
        int c0 = hh * 256;
        float mx = -3.0e38f;
        for (int c = 0; c < 256; ++c) mx = fmaxf(mx, bf2f(P[r][c0 + c]));
        red2[r][hh] = mx;
        __syncthreads();
        mx = fmaxf(red2[r][0], red2[r][1]);
        __syncthreads();
        float sm = 0.f;
        for (int c = 0; c < 256; ++c) {
            float p = __expf(bf2f(P[r][c0 + c]) - mx);
            P[r][c0 + c] = f2bf(p);
            sm += p;
        }
        red2[r][hh] = sm;
        __syncthreads();
        if (hh == 0) rowsum[r] = red2[r][0] + red2[r][1];
        __syncthreads();
    }

    // -------- phase 3: O = P V (normalize by rowsum on the way out) -------
    const unsigned short* Vp = VT + (size_t)hb * ND * NS;   // [D][S]
    f32x8 acc[4] = {};
    for (int kb = 0; kb < NS / 32; ++kb) {
        Frag a;
        const unsigned short* ab = &P[qr0 + l16][kb * 32];
        a.q[0] = *(const u32x4*)(const void*)(ab + hlf * 8);
        a.q[1] = *(const u32x4*)(const void*)(ab + 16 + hlf * 8);
        #pragma unroll
        for (int j = 0; j < 4; ++j) {
            Frag bb;
            const unsigned short* bp = Vp + (size_t)(j * 16 + l16) * NS + kb * 32;
            bb.q[0] = *(const u32x4*)(const void*)(bp + hlf * 8);
            bb.q[1] = *(const u32x4*)(const void*)(bp + 16 + hlf * 8);
            acc[j] = __builtin_amdgcn_wmma_f32_16x16x32_bf16(
                        false, a.v, false, bb.v, (short)0, acc[j], false, false);
        }
    }
    #pragma unroll
    for (int j = 0; j < 4; ++j) {
        #pragma unroll
        for (int v = 0; v < 8; ++v) {
            int qrel = qr0 + hlf * 8 + v;
            int qa   = q0 + qrel;
            int d    = j * 16 + l16;
            float val = acc[j][v] * (1.0f / rowsum[qrel]);
            OUT[((size_t)(b * NS + qa)) * NE + h * ND + d] = val;
        }
    }
}

// ---------------------------------------------------------------------------
// Embedding gather + sinusoidal positional encoding. grid = B*S, block = 256
// ---------------------------------------------------------------------------
__global__ __launch_bounds__(256) void embed_kernel(
    const int* __restrict__ tok, const float* __restrict__ emb,
    float* __restrict__ out)
{
    int idx = blockIdx.x;                 // b*S + s
    int sPos = idx & (NS - 1);
    int t = tok[idx];
    for (int e = threadIdx.x; e < NE; e += 256) {
        int k2 = e & ~1;
        float freq = __expf(-9.210340371976184f * (float)k2 / (float)NE);
        float ang  = (float)sPos * freq;
        float pe   = (e & 1) ? __cosf(ang) : __sinf(ang);
        out[(size_t)idx * NE + e] = emb[(size_t)t * NE + e] + pe;
    }
}

// ---------------------------------------------------------------------------
// Split heads per the reference quirk (channel c = d*H + h) into bf16:
// QH,KH: [B,H,S,D];  VT: [B,H,D,S] (transposed for P*V fragment loads).
// ---------------------------------------------------------------------------
__global__ __launch_bounds__(256) void rearrange_kernel(
    const float* __restrict__ QP, const float* __restrict__ KP,
    const float* __restrict__ VP,
    unsigned short* __restrict__ QH, unsigned short* __restrict__ KH,
    unsigned short* __restrict__ VT)
{
    int i = blockIdx.x * 256 + threadIdx.x;   // over B*S*E = 2,097,152
    int c = i & (NE - 1);
    int s = (i >> 9) & (NS - 1);
    int b = i >> 18;
    int d = c >> 3, h = c & 7;
    int hb = b * NH + h;
    QH[((size_t)(hb * NS + s)) * ND + d] = f2bf(QP[i]);
    KH[((size_t)(hb * NS + s)) * ND + d] = f2bf(KP[i]);
    VT[((size_t)(hb * ND + d)) * NS + s] = f2bf(VP[i]);
}

// ---------------------------------------------------------------------------
// out = LayerNorm(x + r) * g + beta      (row = 512, block 256, 2 elems/thr)
// ---------------------------------------------------------------------------
__global__ __launch_bounds__(256) void add_ln_kernel(
    const float* __restrict__ x, const float* __restrict__ r,
    const float* __restrict__ g, const float* __restrict__ beta,
    float* __restrict__ out)
{
    __shared__ float sh[8];
    int row = blockIdx.x, tid = threadIdx.x;
    const float* xr = x + (size_t)row * NE;
    const float* rr = r + (size_t)row * NE;
    float a0 = xr[tid] + rr[tid];
    float a1 = xr[tid + 256] + rr[tid + 256];
    float mean = block_sum(a0 + a1, sh) * (1.0f / NE);
    float d0 = a0 - mean, d1 = a1 - mean;
    float var = block_sum(d0 * d0 + d1 * d1, sh) * (1.0f / NE);
    float inv = rsqrtf(var + 1e-5f);
    out[(size_t)row * NE + tid]       = d0 * inv * g[tid] + beta[tid];
    out[(size_t)row * NE + tid + 256] = d1 * inv * g[tid + 256] + beta[tid + 256];
}

// grab y[:, -1, :] -> LAST[8,512]
__global__ __launch_bounds__(256) void gather_last_kernel(
    const float* __restrict__ Y, float* __restrict__ L)
{
    int b = blockIdx.x, tid = threadIdx.x;
    L[b * NE + tid]       = Y[((size_t)(b * NS + NS - 1)) * NE + tid];
    L[b * NE + tid + 256] = Y[((size_t)(b * NS + NS - 1)) * NE + tid + 256];
}

// row softmax over N=32000 logits, grid = 8 rows
__global__ __launch_bounds__(256) void softmax_kernel(
    const float* __restrict__ lg, float* __restrict__ out, int N)
{
    __shared__ float sh[8];
    int row = blockIdx.x, tid = threadIdx.x;
    const float* L = lg + (size_t)row * N;
    float mx = -3.0e38f;
    for (int c = tid; c < N; c += 256) mx = fmaxf(mx, L[c]);
    mx = block_max(mx, sh);
    float s = 0.f;
    for (int c = tid; c < N; c += 256) s += __expf(L[c] - mx);
    s = block_sum(s, sh);
    float inv = 1.0f / s;
    float* O = out + (size_t)row * N;
    for (int c = tid; c < N; c += 256) O[c] = __expf(L[c] - mx) * inv;
}

// ---------------------------------------------------------------------------
// Host orchestration
// ---------------------------------------------------------------------------
struct WSPtrs {
    float *X, *Y, *T0, *T1, *QP, *KP, *VP, *AC, *LAST, *LOGITS;
    unsigned short *QH, *KH, *VT, *WB;
};

static void launch_gemm(const float* A, const float* W, const float* bias,
                        float* C, int M, int N, int K, int relu,
                        unsigned short* WB, hipStream_t s)
{
    int n4 = (N * K) / 4;
    cvt_bf16_kernel<<<(n4 + 255) / 256, 256, 0, s>>>(W, WB, n4);
    dim3 grid(N / 128, (M + 127) / 128);
    if ((M & 127) == 0)
        gemm_bias_kernel<false><<<grid, 256, 0, s>>>(A, WB, bias, C, M, N, K, relu);
    else
        gemm_bias_kernel<true><<<grid, 256, 0, s>>>(A, WB, bias, C, M, N, K, relu);
}

static void run_mha(const float* qin, const float* kvin,
                    const float* const p[8],   // qW,qb,kW,kb,vW,vb,oW,ob
                    const WSPtrs& w, float* outBuf, int causal, hipStream_t s)
{
    launch_gemm(qin,  p[0], p[1], w.QP, MTOK, NE, NE, 0, w.WB, s);
    launch_gemm(kvin, p[2], p[3], w.KP, MTOK, NE, NE, 0, w.WB, s);
    launch_gemm(kvin, p[4], p[5], w.VP, MTOK, NE, NE, 0, w.WB, s);
    rearrange_kernel<<<(MTOK * NE) / 256, 256, 0, s>>>(w.QP, w.KP, w.VP,
                                                       w.QH, w.KH, w.VT);
    attn_kernel<<<dim3(NS / QB, NH, NB), 64, 0, s>>>(w.QH, w.KH, w.VT,
                                                     w.AC, causal);
    launch_gemm(w.AC, p[6], p[7], outBuf, MTOK, NE, NE, 0, w.WB, s);
}

extern "C" void kernel_launch(void* const* d_in, const int* in_sizes, int n_in,
                              void* d_out, int out_size, void* d_ws, size_t ws_size,
                              hipStream_t stream)
{
    (void)in_sizes; (void)n_in; (void)out_size; (void)ws_size;

    const int*   toks_in  = (const int*)d_in[0];
    const int*   toks_out = (const int*)d_in[1];
    const float* in_emb   = (const float*)d_in[2];
    const float* out_emb  = (const float*)d_in[3];
    const float* projW    = (const float*)d_in[4];
    const float* projb    = (const float*)d_in[5];
    auto F = [&](int i) { return (const float*)d_in[i]; };

    // workspace carve-out (~139 MB)
    char* ws = (char*)d_ws;
    WSPtrs w;
    size_t off = 0;
    const size_t SZ_ACT = (size_t)MTOK * NE * 4;         // 8 MB
    w.X      = (float*)(ws + off); off += SZ_ACT;
    w.Y      = (float*)(ws + off); off += SZ_ACT;
    w.T0     = (float*)(ws + off); off += SZ_ACT;
    w.T1     = (float*)(ws + off); off += (size_t)MTOK * NFF * 4;   // 32 MB
    w.QP     = (float*)(ws + off); off += SZ_ACT;
    w.KP     = (float*)(ws + off); off += SZ_ACT;
    w.VP     = (float*)(ws + off); off += SZ_ACT;
    w.QH     = (unsigned short*)(ws + off); off += (size_t)MTOK * NE * 2;
    w.KH     = (unsigned short*)(ws + off); off += (size_t)MTOK * NE * 2;
    w.VT     = (unsigned short*)(ws + off); off += (size_t)MTOK * NE * 2;
    w.AC     = (float*)(ws + off); off += SZ_ACT;
    w.LAST   = (float*)(ws + off); off += (size_t)NB * NE * 4;
    w.LOGITS = (float*)(ws + off); off += (size_t)NB * NV * 4;
    w.WB     = (unsigned short*)(ws + off); off += (size_t)NV * NE * 2; // 32 MB

    // ---- encoder ----
    embed_kernel<<<MTOK, 256, 0, stream>>>(toks_in, in_emb, w.X);
    for (int l = 0; l < NL; ++l) {
        int p = 6 + 14 * l;   // qW,qb,kW,kb,vW,vb,oW,ob,fc1W,fc1b,fc2W,fc2b,g,b
        const float* mp[8];
        for (int j = 0; j < 8; ++j) mp[j] = F(p + j);
        run_mha(w.X, w.X, mp, w, w.T0, 0, stream);
        add_ln_kernel<<<MTOK, 256, 0, stream>>>(w.X, w.T0, F(p + 12), F(p + 13), w.X);
        launch_gemm(w.X,  F(p + 8),  F(p + 9),  w.T1, MTOK, NFF, NE,  1, w.WB, stream);
        launch_gemm(w.T1, F(p + 10), F(p + 11), w.T0, MTOK, NE,  NFF, 0, w.WB, stream);
        add_ln_kernel<<<MTOK, 256, 0, stream>>>(w.X, w.T0, F(p + 12), F(p + 13), w.X);
    }
    // w.X now holds encoder memory

    // ---- decoder ----
    embed_kernel<<<MTOK, 256, 0, stream>>>(toks_out, out_emb, w.Y);
    for (int l = 0; l < NL; ++l) {
        int p = 90 + 22 * l;   // mha1(8), mha2(8), ffn(4), ln(2)
        const float *m1[8], *m2[8];
        for (int j = 0; j < 8; ++j) { m1[j] = F(p + j); m2[j] = F(p + 8 + j); }
        run_mha(w.Y, w.Y, m1, w, w.T0, 1, stream);   // masked self-attn
        add_ln_kernel<<<MTOK, 256, 0, stream>>>(w.Y, w.T0, F(p + 20), F(p + 21), w.Y);
        run_mha(w.Y, w.X, m2, w, w.T0, 0, stream);   // cross-attn vs memory
        add_ln_kernel<<<MTOK, 256, 0, stream>>>(w.Y, w.T0, F(p + 20), F(p + 21), w.Y);
        launch_gemm(w.Y,  F(p + 16), F(p + 17), w.T1, MTOK, NFF, NE,  1, w.WB, stream);
        launch_gemm(w.T1, F(p + 18), F(p + 19), w.T0, MTOK, NE,  NFF, 0, w.WB, stream);
        add_ln_kernel<<<MTOK, 256, 0, stream>>>(w.Y, w.T0, F(p + 20), F(p + 21), w.Y);
    }

    // ---- output projection on last target position + softmax ----
    gather_last_kernel<<<NB, 256, 0, stream>>>(w.Y, w.LAST);
    launch_gemm(w.LAST, projW, projb, w.LOGITS, NB, NV, NE, 0, w.WB, stream);
    softmax_kernel<<<NB, 256, 0, stream>>>(w.LOGITS, (float*)d_out, NV);
}
